// SmartMoEFFN_77378130805203
// MI455X (gfx1250) — matrix operand order repaired and checked by reading, not compiled, not run on hardware
//
#include <hip/hip_runtime.h>
#include <math.h>

// ---- problem constants ----
#define TOKENS 8192      // B*S
#define SS     2048      // S
#define DD     1024      // D
#define HH     4096      // H
#define EE     8         // experts

typedef __bf16 bf16_t;
typedef bf16_t v16bf __attribute__((ext_vector_type(16)));
typedef float  v8f   __attribute__((ext_vector_type(8)));
typedef unsigned int u32x4 __attribute__((ext_vector_type(4)));

union FragB { v16bf v; u32x4 q[2]; };

__device__ __forceinline__ unsigned short f2bf(float f) {
  unsigned int u = __float_as_uint(f);
  unsigned int r = u + 0x7fffu + ((u >> 16) & 1u);   // round-to-nearest-even
  return (unsigned short)(r >> 16);
}

__device__ __forceinline__ v8f wmma_bf16(v16bf a, v16bf b, v8f c) {
  // emits v_wmma_f32_16x16x32_bf16
  return __builtin_amdgcn_wmma_f32_16x16x32_bf16(false, a, false, b, (short)0, c, false, false);
}

__device__ __forceinline__ float gelu_exact(float x) {
  return 0.5f * x * (1.0f + erff(x * 0.70710678118654752f));
}

// ---- async copy: 32KB activation tile (global -> LDS) via CDNA5 async path ----
// 256 threads x 8 iters x 16B = 32KB; tracked with ASYNCcnt.
__device__ __forceinline__ void stage_tile_async(unsigned short* sdst,
                                                 const unsigned short* gsrc, int tid) {
  for (int it = 0; it < 8; ++it) {
    int idx = (tid + it * 256) * 8;   // element (ushort) offset, 16B per chunk
    unsigned lds_off = (unsigned)(unsigned long long)(sdst + idx);
    asm volatile("global_load_async_to_lds_b128 %0, %1, off"
                 :: "v"(lds_off), "v"(gsrc + idx) : "memory");
  }
  asm volatile("s_wait_asynccnt 0x0" ::: "memory");
}

// ---------------- aux accumulator zeroing ----------------
__global__ void k_zero_aux(float* aux) {
  if (threadIdx.x < 16) aux[threadIdx.x] = 0.0f;
}

// ---------------- weight convert + swizzle into WMMA B-fragment order ----------------
// dst tile layout: tileIdx=(kt*NT+nt); within tile: [lane 0..31][elem 0..15]
//   column n = nt*16 + (lane&15);  k = kt*32 + (lane>>4)*8 + (elem&7) + (elem>>3)*16   (ISA 16-bit B layout)
__global__ void k_swizzleB(const float* __restrict__ src, unsigned short* __restrict__ dst,
                           int K, int N, long long total) {
  long long idx = (long long)blockIdx.x * blockDim.x + threadIdx.x;
  if (idx >= total) return;
  long long pb = (long long)K * N;
  long long b  = idx / pb;
  long long rem = idx - b * pb;
  int i    = (int)(rem & 15);
  int lane = (int)((rem >> 4) & 31);
  long long tile = rem >> 9;
  int NT = N >> 4;
  int nt = (int)(tile % NT);
  int kt = (int)(tile / NT);
  int k = kt * 32 + (lane >> 4) * 8 + (i & 7) + (i >> 3) * 16;
  int n = nt * 16 + (lane & 15);
  dst[idx] = f2bf(src[(size_t)b * pb + (size_t)k * N + n]);
}

// ---------------- LayerNorm(x) -> rf base (+quality term), and x -> bf16 ----------------
__global__ __launch_bounds__(256) void k_ln_x(const float* __restrict__ x,
    const float* __restrict__ g, const float* __restrict__ bb,
    const float* __restrict__ quality, const float* __restrict__ qW,
    const float* __restrict__ qb, float* __restrict__ rf,
    unsigned short* __restrict__ xbf) {
  __shared__ float s1[256], s2[256];
  const int row = blockIdx.x, tid = threadIdx.x;
  const float* xr = x + (size_t)row * DD;
  float a = 0.f, a2 = 0.f;
  for (int it = 0; it < 4; ++it) { float v = xr[tid + it * 256]; a += v; a2 += v * v; }
  s1[tid] = a; s2[tid] = a2; __syncthreads();
  for (int st = 128; st > 0; st >>= 1) {
    if (tid < st) { s1[tid] += s1[tid + st]; s2[tid] += s2[tid + st]; }
    __syncthreads();
  }
  float mu = s1[0] * (1.0f / DD);
  float var = s2[0] * (1.0f / DD) - mu * mu;
  float rst = rsqrtf(var + 1e-5f);
  float qv = quality[row / SS];
  for (int it = 0; it < 4; ++it) {
    int c = tid + it * 256;
    float xv = xr[c];
    float ln = (xv - mu) * rst * g[c] + bb[c];
    rf[(size_t)row * DD + c] = ln + qv * qW[c] + qb[c];
    xbf[(size_t)row * DD + c] = f2bf(xv);
  }
}

// ---------------- LayerNorm(context) -> bf16 ----------------
__global__ __launch_bounds__(256) void k_ln_ctx(const float* __restrict__ ctx,
    const float* __restrict__ g, const float* __restrict__ bb,
    unsigned short* __restrict__ cnbf) {
  __shared__ float s1[256], s2[256];
  const int row = blockIdx.x, tid = threadIdx.x;
  const float* xr = ctx + (size_t)row * DD;
  float a = 0.f, a2 = 0.f;
  for (int it = 0; it < 4; ++it) { float v = xr[tid + it * 256]; a += v; a2 += v * v; }
  s1[tid] = a; s2[tid] = a2; __syncthreads();
  for (int st = 128; st > 0; st >>= 1) {
    if (tid < st) { s1[tid] += s1[tid + st]; s2[tid] += s2[tid + st]; }
    __syncthreads();
  }
  float mu = s1[0] * (1.0f / DD);
  float var = s2[0] * (1.0f / DD) - mu * mu;
  float rst = rsqrtf(var + 1e-5f);
  for (int it = 0; it < 4; ++it) {
    int c = tid + it * 256;
    float ln = (xr[c] - mu) * rst * g[c] + bb[c];
    cnbf[(size_t)row * DD + c] = f2bf(ln);
  }
}

// ---------------- ctx GEMM: rf += LN(context) @ ctx_W + ctx_b  (bf16 WMMA) ----------------
__global__ __launch_bounds__(256) void k_ctx_gemm(const unsigned short* __restrict__ cnbf,
    const unsigned short* __restrict__ cwbf, const float* __restrict__ ctx_b,
    float* __restrict__ rf) {
  __shared__ __align__(16) unsigned short sA[16 * DD]; // 32KB: 16 tokens x 1024, row-major bf16
  const int tid = threadIdx.x, tile = blockIdx.x;
  stage_tile_async(sA, cnbf + (size_t)tile * 16 * DD, tid);
  __syncthreads();
  const int w = tid >> 5, lane = tid & 31, m = lane & 15, half = lane >> 4;
  v8f acc[8];
#pragma unroll
  for (int i = 0; i < 8; ++i) acc[i] = {0.f,0.f,0.f,0.f,0.f,0.f,0.f,0.f};
  for (int kt = 0; kt < 32; ++kt) {
    FragB a;
    const unsigned short* ap = sA + m * DD + kt * 32 + half * 8;
    a.q[0] = *(const u32x4*)ap;
    a.q[1] = *(const u32x4*)(ap + 16);
#pragma unroll
    for (int nt = 0; nt < 8; ++nt) {
      int gnt = w * 8 + nt;                       // N-tile in 0..63
      FragB b;
      const u32x4* bp = (const u32x4*)(cwbf + ((size_t)(kt * 64 + gnt)) * 512 + lane * 16);
      b.q[0] = bp[0]; b.q[1] = bp[1];
      acc[nt] = wmma_bf16(a.v, b.v, acc[nt]);
    }
  }
#pragma unroll
  for (int nt = 0; nt < 8; ++nt) {
    int col = (w * 8 + nt) * 16 + (lane & 15);
    float cbv = ctx_b[col];
#pragma unroll
    for (int r = 0; r < 8; ++r) {
      int row = half * 8 + r;                     // ISA C/D layout
      size_t idx = ((size_t)(tile * 16 + row)) * DD + col;
      rf[idx] += acc[nt][r] + cbv;
    }
  }
}

// ---------------- router: logits, top-2 softmax -> combine, aux partials ----------------
__global__ __launch_bounds__(128) void k_router(const float* __restrict__ rf,
    const float* __restrict__ rW, const float* __restrict__ rb,
    const float* __restrict__ temp, float* __restrict__ combine,
    float* __restrict__ aux) {
  __shared__ float srw[DD * EE];   // 32KB
  __shared__ float sp[EE * 128];
  const int row = blockIdx.x, tid = threadIdx.x;
  for (int it = 0; it < 64; ++it) { int i = tid + it * 128; srw[i] = rW[i]; }
  __syncthreads();
  float p[EE];
#pragma unroll
  for (int e = 0; e < EE; ++e) p[e] = 0.f;
  for (int it = 0; it < 8; ++it) {
    int i = tid + it * 128;
    float rv = rf[(size_t)row * DD + i];
#pragma unroll
    for (int e = 0; e < EE; ++e) p[e] += rv * srw[i * EE + e];
  }
#pragma unroll
  for (int e = 0; e < EE; ++e) sp[e * 128 + tid] = p[e];
  __syncthreads();
  for (int st = 64; st > 0; st >>= 1) {
    if (tid < st) {
#pragma unroll
      for (int e = 0; e < EE; ++e) sp[e * 128 + tid] += sp[e * 128 + tid + st];
    }
    __syncthreads();
  }
  if (tid == 0) {
    float t = fmaxf(temp[0], 0.25f);
    float lg[EE];
#pragma unroll
    for (int e = 0; e < EE; ++e) lg[e] = (sp[e * 128] + rb[e]) / t;
    int i0 = 0;
#pragma unroll
    for (int e = 1; e < EE; ++e) if (lg[e] > lg[i0]) i0 = e;
    int i1 = (i0 == 0) ? 1 : 0;
#pragma unroll
    for (int e = 0; e < EE; ++e) if (e != i0 && lg[e] > lg[i1]) i1 = e;
    float e1 = expf(lg[i1] - lg[i0]);
    float inv = 1.0f / (1.0f + e1);
#pragma unroll
    for (int e = 0; e < EE; ++e) combine[(size_t)row * EE + e] = 0.f;
    combine[(size_t)row * EE + i0] = inv;
    combine[(size_t)row * EE + i1] = e1 * inv;
    // full softmax for aux losses
    float mx = lg[i0], se = 0.f;
#pragma unroll
    for (int e = 0; e < EE; ++e) se += expf(lg[e] - mx);
    float lse = mx + logf(se), ent = 0.f;
#pragma unroll
    for (int e = 0; e < EE; ++e) {
      float pe = expf(lg[e] - mx) / se;
      atomicAdd(&aux[e], pe);
      ent -= pe * logf(fmaxf(pe, 1e-9f));
    }
    atomicAdd(&aux[8], lse * lse);
    atomicAdd(&aux[9], ent);
  }
}

__global__ void k_finalize(const float* __restrict__ aux, float* __restrict__ tail) {
  if (threadIdx.x != 0 || blockIdx.x != 0) return;
  const float inv = 1.0f / (float)TOKENS;
  float lb = 0.f;
  for (int e = 0; e < EE; ++e) { float d = aux[e] * inv - (1.0f / EE); lb += d * d; }
  lb *= (1.0f / EE);
  float rz = aux[8] * inv, ent = aux[9] * inv;
  tail[0] = lb; tail[1] = rz; tail[2] = ent;
  tail[3] = lb + 0.001f * rz - 0.001f * ent;
}

// ---------------- fused MoE FFN: out = sum_e combine_e * (gelu(x@W1_e+b1)@W2_e + b2) ----------------
__global__ __launch_bounds__(256) void k_ffn(const unsigned short* __restrict__ xbf,
    const unsigned short* __restrict__ W1bf, const unsigned short* __restrict__ W2bf,
    const float* __restrict__ b1, const float* __restrict__ b2,
    const float* __restrict__ combine, float* __restrict__ out) {
  __shared__ __align__(16) unsigned short sX[16 * DD];      // 32KB token tile (bf16, row-major)
  __shared__ __align__(16) unsigned short sH[2][16 * 128];  // 2x4KB ping-pong h-block
  __shared__ float sC[16 * EE];
  __shared__ float sFlag[EE];
  const int tid = threadIdx.x, tile = blockIdx.x;
  stage_tile_async(sX, xbf + (size_t)tile * 16 * DD, tid);
  if (tid < 128) sC[tid] = combine[(size_t)(tile * 16 + (tid >> 3)) * EE + (tid & 7)];
  __syncthreads();
  if (tid < EE) {
    float mx = 0.f;
    for (int r = 0; r < 16; ++r) mx = fmaxf(mx, sC[r * EE + tid]);
    sFlag[tid] = mx;
  }
  __syncthreads();

  const int w = tid >> 5, lane = tid & 31, m = lane & 15, half = lane >> 4;
  const v8f vz = {0.f,0.f,0.f,0.f,0.f,0.f,0.f,0.f};
  v8f oacc[8];
#pragma unroll
  for (int i = 0; i < 8; ++i) oacc[i] = vz;

  for (int e = 0; e < EE; ++e) {
    if (!(sFlag[e] > 0.f)) continue;               // uniform branch (shared value)
    v8f yacc[8];
#pragma unroll
    for (int i = 0; i < 8; ++i) yacc[i] = vz;
    const size_t w1base = (size_t)e * (32 * 256) * 512;   // tiles(kt=32,nt=256) * 512 elems
    const size_t w2base = (size_t)e * (128 * 64) * 512;   // tiles(kt=128,nt=64) * 512 elems

    for (int hb = 0; hb < HH / 128; ++hb) {
      unsigned short* hbuf = sH[hb & 1];
      // ---- GEMM1: this wave's 16-col slice of the 16x128 h-block ----
      // two accumulators break the serial WMMA RAW chain (low occupancy: ~200 VGPR/lane)
      v8f hacc0 = vz, hacc1 = vz;
      const int nt1 = hb * 8 + w;                  // W1 N-tile (H dim)
      for (int kt = 0; kt < 32; kt += 2) {
        FragB a0, a1, b0, b1f;
        const unsigned short* ap = sX + m * DD + kt * 32 + half * 8;
        a0.q[0] = *(const u32x4*)ap;
        a0.q[1] = *(const u32x4*)(ap + 16);
        a1.q[0] = *(const u32x4*)(ap + 32);
        a1.q[1] = *(const u32x4*)(ap + 48);
        const u32x4* bp = (const u32x4*)(W1bf + w1base + ((size_t)(kt * 256 + nt1)) * 512 + lane * 16);
        __builtin_prefetch(bp + 512, 0, 1);        // 8 tiles (8KB) ahead in packed W1
        b0.q[0] = bp[0]; b0.q[1] = bp[1];
        const u32x4* bp1 = bp + 256 * 32;          // 8192 u32x4 = 128 tiles ahead of bp, not next kt (256 tiles)
        b1f.q[0] = bp1[0]; b1f.q[1] = bp1[1];
        hacc0 = wmma_bf16(a0.v, b0.v, hacc0);
        hacc1 = wmma_bf16(a1.v, b1f.v, hacc1);
      }
      const int colH = hb * 128 + w * 16 + (lane & 15);
      const float b1v = b1[e * HH + colH];
#pragma unroll
      for (int r = 0; r < 8; ++r) {
        float hv = gelu_exact(hacc0[r] + hacc1[r] + b1v);
        hbuf[(half * 8 + r) * 128 + w * 16 + (lane & 15)] = f2bf(hv);
      }
      // single barrier per h-block: ping-pong buffer makes the write->read and
      // read->rewrite hazards both covered by this one barrier per iteration.
      __syncthreads();
      // ---- GEMM2: all waves consume the 16x128 h-block ----
      for (int kt2 = 0; kt2 < 4; ++kt2) {
        FragB a2;
        const unsigned short* hp = hbuf + m * 128 + kt2 * 32 + half * 8;
        a2.q[0] = *(const u32x4*)hp;
        a2.q[1] = *(const u32x4*)(hp + 16);
#pragma unroll
        for (int nt = 0; nt < 8; ++nt) {
          FragB b2f;
          const u32x4* bp2 = (const u32x4*)(W2bf + w2base +
              ((size_t)((hb * 4 + kt2) * 64 + (w * 8 + nt))) * 512 + lane * 16);
          __builtin_prefetch(bp2 + 64 * 32, 0, 1); // 32 tiles (32KB) ahead in packed W2
          b2f.q[0] = bp2[0]; b2f.q[1] = bp2[1];
          yacc[nt] = wmma_bf16(a2.v, b2f.v, yacc[nt]);
        }
      }
    }
    // ---- scale by per-token gate, add expert bias ----
#pragma unroll
    for (int nt = 0; nt < 8; ++nt) {
      int colD = (w * 8 + nt) * 16 + (lane & 15);
      float b2v = b2[e * DD + colD];
#pragma unroll
      for (int r = 0; r < 8; ++r) {
        int row = half * 8 + r;
        float ce = sC[row * EE + e];
        oacc[nt][r] += ce * (yacc[nt][r] + b2v);
      }
    }
  }
#pragma unroll
  for (int nt = 0; nt < 8; ++nt) {
    int colD = (w * 8 + nt) * 16 + (lane & 15);
#pragma unroll
    for (int r = 0; r < 8; ++r) {
      int row = half * 8 + r;
      out[((size_t)(tile * 16 + row)) * DD + colD] = oacc[nt][r];
    }
  }
}

// ---------------- host launch ----------------
extern "C" void kernel_launch(void* const* d_in, const int* in_sizes, int n_in,
                              void* d_out, int out_size, void* d_ws, size_t ws_size,
                              hipStream_t stream) {
  (void)in_sizes; (void)n_in; (void)out_size; (void)ws_size;
  const float* x        = (const float*)d_in[0];
  const float* context  = (const float*)d_in[1];
  const float* quality  = (const float*)d_in[2];
  const float* rn_g     = (const float*)d_in[3];
  const float* rn_b     = (const float*)d_in[4];
  const float* cn_g     = (const float*)d_in[5];
  const float* cn_b     = (const float*)d_in[6];
  const float* ctx_W    = (const float*)d_in[7];
  const float* ctx_b    = (const float*)d_in[8];
  const float* q_W      = (const float*)d_in[9];
  const float* q_b      = (const float*)d_in[10];
  const float* r_W      = (const float*)d_in[11];
  const float* r_b      = (const float*)d_in[12];
  const float* temp     = (const float*)d_in[13];
  const float* W1       = (const float*)d_in[14];
  const float* b1       = (const float*)d_in[15];
  const float* W2       = (const float*)d_in[16];
  const float* b2       = (const float*)d_in[17];

  char* ws = (char*)d_ws;
  size_t o = 0;
  unsigned short* xbf  = (unsigned short*)(ws + o); o += (size_t)TOKENS * DD * 2;
  unsigned short* cnbf = (unsigned short*)(ws + o); o += (size_t)TOKENS * DD * 2;
  float*          rf   = (float*)(ws + o);          o += (size_t)TOKENS * DD * 4;
  unsigned short* cwbf = (unsigned short*)(ws + o); o += (size_t)DD * DD * 2;
  unsigned short* w1bf = (unsigned short*)(ws + o); o += (size_t)EE * DD * HH * 2;
  unsigned short* w2bf = (unsigned short*)(ws + o); o += (size_t)EE * HH * DD * 2;
  float*          comb = (float*)(ws + o);          o += (size_t)TOKENS * EE * 4;
  float*          aux  = (float*)(ws + o);          o += 64;

  float* outp = (float*)d_out;
  float* tail = outp + (size_t)TOKENS * DD;

  k_zero_aux<<<1, 32, 0, stream>>>(aux);

  {
    long long t0 = (long long)DD * DD;
    k_swizzleB<<<(unsigned)((t0 + 255) / 256), 256, 0, stream>>>(ctx_W, cwbf, DD, DD, t0);
    long long t1 = (long long)EE * DD * HH;
    k_swizzleB<<<(unsigned)((t1 + 255) / 256), 256, 0, stream>>>(W1, w1bf, DD, HH, t1);
    long long t2 = (long long)EE * HH * DD;
    k_swizzleB<<<(unsigned)((t2 + 255) / 256), 256, 0, stream>>>(W2, w2bf, HH, DD, t2);
  }

  k_ln_x  <<<TOKENS, 256, 0, stream>>>(x, rn_g, rn_b, quality, q_W, q_b, rf, xbf);
  k_ln_ctx<<<TOKENS, 256, 0, stream>>>(context, cn_g, cn_b, cnbf);
  k_ctx_gemm<<<TOKENS / 16, 256, 0, stream>>>(cnbf, cwbf, ctx_b, rf);
  k_router<<<TOKENS, 128, 0, stream>>>(rf, r_W, r_b, temp, comb, aux);
  k_finalize<<<1, 1, 0, stream>>>(aux, tail);
  k_ffn<<<TOKENS / 16, 256, 0, stream>>>(xbf, w1bf, w2bf, b1, b2, comb, outp);
}
